// GmapAD_18743237280517
// MI455X (gfx1250) — compile-verified, matrix operand order checked by reading, not run on hardware
//
#include <hip/hip_runtime.h>
#include <hip/hip_bf16.h>

#define N_NODES  10000
#define N_EDGES  640000
#define N_GRAPHS 100
#define IN_DIM   128
#define HID_DIM  128
#define OUT_DIM  64
#define N_CAND   64

typedef __attribute__((ext_vector_type(16))) __bf16 v16bf;
typedef __attribute__((ext_vector_type(8)))  __bf16 v8bf;
typedef __attribute__((ext_vector_type(8)))  float  v8f;

// ---------------- utility kernels ----------------

__global__ void k_zero(float* p, int n) {
    int t = blockIdx.x * blockDim.x + threadIdx.x;
    if (t < n) p[t] = 0.0f;
}

__global__ void k_degree(const int* __restrict__ dst, float* __restrict__ deg, int nE) {
    int t = blockIdx.x * blockDim.x + threadIdx.x;
    if (t < nE) atomicAdd(&deg[dst[t]], 1.0f);
}

__global__ void k_dinv(const float* __restrict__ deg, float* __restrict__ dinv, int n) {
    int t = blockIdx.x * blockDim.x + threadIdx.x;
    if (t < n) dinv[t] = rsqrtf(deg[t] + 1.0f);
}

__global__ void k_cvt_bf16(const float* __restrict__ in, __bf16* __restrict__ out, int n) {
    int t = blockIdx.x * blockDim.x + threadIdx.x;
    if (t < n) out[t] = (__bf16)in[t];
}

// W [rows, cols] row-major  ->  Wt [cols, rows] bf16 (so B fragments load contiguously over K)
__global__ void k_transpose_bf16(const float* __restrict__ W, __bf16* __restrict__ Wt,
                                 int rows, int cols) {
    int t = blockIdx.x * blockDim.x + threadIdx.x;
    if (t >= rows * cols) return;
    int r = t / cols, c = t % cols;
    Wt[c * rows + r] = (__bf16)W[r * cols + c];
}

// ---------------- WMMA GEMM: D[M,N] = A[M,K](bf16) * Bt[N,K](bf16), f32 accum ----------------
// One wave computes one 16x16 tile. Block has N/16 waves (one M-tile row per block).
// A-fragment layout (16-bit A, 16x32): lane holds row M=lane&15, kbase=(lane>>4)*8,
// element i -> K = (i>>3)*16 + kbase + (i&7). Both halves are contiguous 16B loads.
template<int K, int N>
__global__ void k_gemm_wmma(const __bf16* __restrict__ A,
                            const __bf16* __restrict__ Bt,
                            float* __restrict__ D, int M) {
    const int lane  = threadIdx.x & 31;
    const int wave  = threadIdx.x >> 5;
    const int mtile = blockIdx.x;
    const int ntile = wave;                 // N/16 waves per block
    const int row   = mtile * 16 + (lane & 15);
    const int col   = ntile * 16 + (lane & 15);
    const int khalf = (lane >> 4) * 8;

    v8f c = {};
#pragma unroll
    for (int kc = 0; kc < K; kc += 32) {
        const __bf16* pa = A  + (long long)row * K + kc + khalf;
        const __bf16* pb = Bt + (long long)col * K + kc + khalf;
        v8bf a0 = *(const v8bf*)(pa);
        v8bf a1 = *(const v8bf*)(pa + 16);
        v8bf b0 = *(const v8bf*)(pb);
        v8bf b1 = *(const v8bf*)(pb + 16);
        v16bf av = __builtin_shufflevector(a0, a1, 0,1,2,3,4,5,6,7,8,9,10,11,12,13,14,15);
        v16bf bv = __builtin_shufflevector(b0, b1, 0,1,2,3,4,5,6,7,8,9,10,11,12,13,14,15);
        c = __builtin_amdgcn_wmma_f32_16x16x32_bf16(false, av, false, bv,
                                                    (short)0, c, false, false);
    }
    // D layout: lane gives col=lane&15; VGPR r -> row = (lane>>4)*8 + r
    const int drow = mtile * 16 + (lane >> 4) * 8;
#pragma unroll
    for (int r = 0; r < 8; ++r)
        D[(long long)(drow + r) * N + col] = c[r];
}

// agg[n,d] = h[n,d]*dinv[n]^2 + b[d]   (self-loop contribution + bias)
__global__ void k_agg_init(const float* __restrict__ h, const float* __restrict__ dinv,
                           const float* __restrict__ b, float* __restrict__ agg,
                           int n, int F) {
    long long t = (long long)blockIdx.x * blockDim.x + threadIdx.x;
    if (t >= (long long)n * F) return;
    int node = (int)(t / F), d = (int)(t % F);
    float di = dinv[node];
    agg[t] = h[t] * di * di + b[d];
}

// agg[dst] += h[src] * dinv[src]*dinv[dst]   per (edge, dim)
__global__ void k_edge_scatter(const float* __restrict__ h, const int* __restrict__ src,
                               const int* __restrict__ dst, const float* __restrict__ dinv,
                               float* __restrict__ agg, int nE, int F) {
    long long t = (long long)blockIdx.x * blockDim.x + threadIdx.x;
    if (t >= (long long)nE * F) return;
    int e = (int)(t / F), d = (int)(t % F);
    int s = src[e], q = dst[e];
    float coef = dinv[s] * dinv[q];
    atomicAdd(&agg[(long long)q * F + d], h[(long long)s * F + d] * coef);
}

__global__ void k_relu_bf16(const float* __restrict__ in, __bf16* __restrict__ out, int n) {
    int t = blockIdx.x * blockDim.x + threadIdx.x;
    if (t < n) out[t] = (__bf16)fmaxf(in[t], 0.0f);
}

__global__ void k_pool(const float* __restrict__ z, const int* __restrict__ batch,
                       float* __restrict__ sums, float* __restrict__ cnt, int n, int F) {
    long long t = (long long)blockIdx.x * blockDim.x + threadIdx.x;
    if (t >= (long long)n * F) return;
    int node = (int)(t / F), d = (int)(t % F);
    int g = batch[node];
    atomicAdd(&sums[g * F + d], z[(long long)node * F + d]);
    if (d == 0) atomicAdd(&cnt[g], 1.0f);
}

__global__ void k_sim(const float* __restrict__ sums, const float* __restrict__ cnt,
                      const float* __restrict__ cand, float* __restrict__ out,
                      int G, int C, int F) {
    int t = blockIdx.x * blockDim.x + threadIdx.x;
    if (t >= G * C) return;
    int g = t / C, c = t % C;
    float inv = 1.0f / fmaxf(cnt[g], 1.0f);
    float acc = 0.0f;
#pragma unroll 8
    for (int d = 0; d < F; ++d) {
        float diff = sums[g * F + d] * inv - cand[c * F + d];
        acc += diff * diff;
    }
    out[t] = sqrtf(acc);
}

// ---------------- launcher ----------------

extern "C" void kernel_launch(void* const* d_in, const int* in_sizes, int n_in,
                              void* d_out, int out_size, void* d_ws, size_t ws_size,
                              hipStream_t stream) {
    const float* x     = (const float*)d_in[0];
    const int*   src   = (const int*)  d_in[1];
    const int*   dst   = (const int*)  d_in[2];
    const int*   batch = (const int*)  d_in[3];
    const float* W1    = (const float*)d_in[4];
    const float* b1    = (const float*)d_in[5];
    const float* W2    = (const float*)d_in[6];
    const float* b2    = (const float*)d_in[7];
    const float* cand  = (const float*)d_in[8];
    float* out = (float*)d_out;

    // workspace carve-out (256B aligned)
    char* base = (char*)d_ws;
    size_t off = 0;
    auto carve = [&](size_t bytes) -> char* {
        char* p = base + off;
        off = (off + bytes + 255) & ~(size_t)255;
        return p;
    };
    float*  deg   = (float*) carve(N_NODES * sizeof(float));
    float*  dinv  = (float*) carve(N_NODES * sizeof(float));
    __bf16* xbf   = (__bf16*)carve((size_t)N_NODES * IN_DIM * 2);
    __bf16* w1t   = (__bf16*)carve((size_t)IN_DIM * HID_DIM * 2);   // [HID, IN]
    __bf16* w2t   = (__bf16*)carve((size_t)HID_DIM * OUT_DIM * 2);  // [OUT, HID]
    float*  h1    = (float*) carve((size_t)N_NODES * HID_DIM * sizeof(float));
    float*  agg1  = (float*) carve((size_t)N_NODES * HID_DIM * sizeof(float));
    __bf16* hrelu = (__bf16*)carve((size_t)N_NODES * HID_DIM * 2);
    float*  h2    = (float*) carve((size_t)N_NODES * OUT_DIM * sizeof(float));
    float*  agg2  = (float*) carve((size_t)N_NODES * OUT_DIM * sizeof(float));
    float*  sums  = (float*) carve((size_t)N_GRAPHS * OUT_DIM * sizeof(float));
    float*  cnt   = (float*) carve((size_t)N_GRAPHS * sizeof(float));
    (void)ws_size;

    const int B = 256;
    auto grid = [](long long n, int b) { return (unsigned)((n + b - 1) / b); };

    // degrees + normalization
    k_zero<<<grid(N_NODES, B), B, 0, stream>>>(deg, N_NODES);
    k_zero<<<grid(N_GRAPHS * OUT_DIM, B), B, 0, stream>>>(sums, N_GRAPHS * OUT_DIM);
    k_zero<<<grid(N_GRAPHS, B), B, 0, stream>>>(cnt, N_GRAPHS);
    k_degree<<<grid(N_EDGES, B), B, 0, stream>>>(dst, deg, N_EDGES);
    k_dinv<<<grid(N_NODES, B), B, 0, stream>>>(deg, dinv, N_NODES);

    // bf16 conversions (weights transposed so WMMA B fragments are contiguous over K)
    k_cvt_bf16<<<grid((long long)N_NODES * IN_DIM, B), B, 0, stream>>>(x, xbf, N_NODES * IN_DIM);
    k_transpose_bf16<<<grid(IN_DIM * HID_DIM, B), B, 0, stream>>>(W1, w1t, IN_DIM, HID_DIM);
    k_transpose_bf16<<<grid(HID_DIM * OUT_DIM, B), B, 0, stream>>>(W2, w2t, HID_DIM, OUT_DIM);

    // layer 1: h1 = x @ W1 via WMMA (625 M-tiles, 8 N-tile waves/block)
    k_gemm_wmma<IN_DIM, HID_DIM><<<N_NODES / 16, (HID_DIM / 16) * 32, 0, stream>>>(xbf, w1t, h1, N_NODES);
    k_agg_init<<<grid((long long)N_NODES * HID_DIM, B), B, 0, stream>>>(h1, dinv, b1, agg1, N_NODES, HID_DIM);
    k_edge_scatter<<<grid((long long)N_EDGES * HID_DIM, B), B, 0, stream>>>(h1, src, dst, dinv, agg1, N_EDGES, HID_DIM);
    k_relu_bf16<<<grid((long long)N_NODES * HID_DIM, B), B, 0, stream>>>(agg1, hrelu, N_NODES * HID_DIM);

    // layer 2: h2 = relu(agg1) @ W2 via WMMA (625 M-tiles, 4 N-tile waves/block)
    k_gemm_wmma<HID_DIM, OUT_DIM><<<N_NODES / 16, (OUT_DIM / 16) * 32, 0, stream>>>(hrelu, w2t, h2, N_NODES);
    k_agg_init<<<grid((long long)N_NODES * OUT_DIM, B), B, 0, stream>>>(h2, dinv, b2, agg2, N_NODES, OUT_DIM);
    k_edge_scatter<<<grid((long long)N_EDGES * OUT_DIM, B), B, 0, stream>>>(h2, src, dst, dinv, agg2, N_EDGES, OUT_DIM);

    // pooling + similarity
    k_pool<<<grid((long long)N_NODES * OUT_DIM, B), B, 0, stream>>>(agg2, batch, sums, cnt, N_NODES, OUT_DIM);
    k_sim<<<grid(N_GRAPHS * N_CAND, B), B, 0, stream>>>(sums, cnt, cand, out, N_GRAPHS, N_CAND, OUT_DIM);
    (void)in_sizes; (void)n_in; (void)out_size;
}
